// ACM_GNN_52012053954566
// MI455X (gfx1250) — matrix operand order repaired and checked by reading, not compiled, not run on hardware
//
#include <hip/hip_runtime.h>

#define DIN 256
#define DOUT 64

typedef __attribute__((ext_vector_type(16))) _Float16     v16h;
typedef __attribute__((ext_vector_type(8)))  float        v8f;
typedef __attribute__((ext_vector_type(8)))  unsigned int v8u;
typedef __attribute__((ext_vector_type(4)))  float        v4f;

static __device__ __forceinline__ unsigned int pack2h(_Float16 a, _Float16 b) {
  union { _Float16 h[2]; unsigned int u; } s;
  s.h[0] = a; s.h[1] = b; return s.u;
}

static __device__ __forceinline__ float wave_sum(float v) {
  #pragma unroll
  for (int off = 16; off > 0; off >>= 1) v += __shfl_xor(v, off, 32);
  return v;
}
static __device__ __forceinline__ float wave_max(float v) {
  #pragma unroll
  for (int off = 16; off > 0; off >>= 1) v = fmaxf(v, __shfl_xor(v, off, 32));
  return v;
}

// ---------------- init: deg=1 (self loop), zero both accumulators ----------
__global__ void k_init(float* __restrict__ deg, float* __restrict__ acc_hp,
                       float* __restrict__ acc_lp, int n) {
  long long t = (long long)blockIdx.x * blockDim.x + threadIdx.x;
  long long tot = (long long)n * DOUT;
  if (t < tot) {
    acc_hp[t] = 0.0f;
    acc_lp[t] = 0.0f;
    if (t < n) deg[t] = 1.0f;
  }
}

// ---------------- degree accumulation over edges ---------------------------
__global__ void k_deg(const int* __restrict__ dst, float* __restrict__ deg, int e) {
  int t = blockIdx.x * blockDim.x + threadIdx.x;
  if (t < e) atomicAdd(&deg[dst[t]], 1.0f);
}

// ---------------- dinv = rsqrt(deg) ----------------------------------------
__global__ void k_dinv(const float* __restrict__ deg, float* __restrict__ dinv, int n) {
  int t = blockIdx.x * blockDim.x + threadIdx.x;
  if (t < n) dinv[t] = rsqrtf(deg[t]);
}

// ---------------- fused triple GEMM: xw_m = X @ W_m + b_m  (WMMA f16 hi/lo) -
// Wave per 16-row tile; block = 8 waves = 128 rows. Per matrix, block packs
// W into LDS as pre-packed B-fragment words [hi/lo][kc][nt][v][lane].
__global__ void __launch_bounds__(256)
k_gemm(const float* __restrict__ x,
       const float* __restrict__ W0, const float* __restrict__ b0,
       const float* __restrict__ W1, const float* __restrict__ b1,
       const float* __restrict__ W2, const float* __restrict__ b2,
       float* __restrict__ o0, float* __restrict__ o1, float* __restrict__ o2,
       int n, int numTiles)
{
  __shared__ unsigned int ldsW[2][8][4][8][32];   // 64 KB

  const int tid  = threadIdx.x;
  const int lane = tid & 31;
  const int wave = tid >> 5;
  const int tile = blockIdx.x * 8 + wave;

  const float* Ws[3]   = {W0, W1, W2};
  const float* bs[3]   = {b0, b1, b2};
  float*       outs[3] = {o0, o1, o2};

  #pragma unroll 1
  for (int m = 0; m < 3; ++m) {
    __syncthreads();
    // ---- pack W[m] (256x64 f32) into hi/lo f16 fragment words ----
    for (int widx = tid; widx < 8192; widx += 256) {
      int l  = widx & 31;
      int v  = (widx >> 5) & 7;
      int nt = (widx >> 8) & 3;
      int kc = widx >> 10;
      int col = nt * 16 + (l & 15);
      int k0  = kc * 32 + ((l & 16) >> 1) + 2 * v + (v >= 4 ? 8 : 0);   // lanes 0-15: K 0..7,16..23, lanes 16-31: K 8..15,24..31
      float w0f = Ws[m][(size_t)k0 * DOUT + col];
      float w1f = Ws[m][(size_t)(k0 + 1) * DOUT + col];
      _Float16 h0 = (_Float16)w0f, h1 = (_Float16)w1f;
      _Float16 g0 = (_Float16)(w0f - (float)h0);
      _Float16 g1 = (_Float16)(w1f - (float)h1);
      ldsW[0][kc][nt][v][l] = pack2h(h0, h1);
      ldsW[1][kc][nt][v][l] = pack2h(g0, g1);
    }
    __syncthreads();

    if (tile < numTiles) {
      v8f acc[4];
      #pragma unroll
      for (int nt = 0; nt < 4; ++nt)
        #pragma unroll
        for (int i = 0; i < 8; ++i) acc[nt][i] = 0.0f;

      int rowA = tile * 16 + (lane & 15);
      if (rowA >= n) rowA = n - 1;                   // clamp (N multiple of 16 anyway)
      const int khalf = (lane & 16) ? 8 : 0;         // lanes 16-31 take K +8 / +24

      #pragma unroll 1
      for (int kc = 0; kc < 8; ++kc) {
        const float* xr = x + (size_t)rowA * DIN + kc * 32 + khalf;
        v4f a0 = *(const v4f*)(xr);
        v4f a1 = *(const v4f*)(xr + 4);
        v4f a2 = *(const v4f*)(xr + 16);
        v4f a3 = *(const v4f*)(xr + 20);
        float vals[16];
        #pragma unroll
        for (int i = 0; i < 4; ++i) {
          vals[i] = a0[i]; vals[4 + i] = a1[i]; vals[8 + i] = a2[i]; vals[12 + i] = a3[i];
        }
        v16h ahi, alo;
        #pragma unroll
        for (int i = 0; i < 16; ++i) {
          _Float16 h = (_Float16)vals[i];
          ahi[i] = h;
          alo[i] = (_Float16)(vals[i] - (float)h);
        }
        #pragma unroll
        for (int nt = 0; nt < 4; ++nt) {
          v8u bhu, blu;
          #pragma unroll
          for (int v = 0; v < 8; ++v) {
            bhu[v] = ldsW[0][kc][nt][v][lane];
            blu[v] = ldsW[1][kc][nt][v][lane];
          }
          v16h bhi = __builtin_bit_cast(v16h, bhu);
          v16h blo = __builtin_bit_cast(v16h, blu);
          acc[nt] = __builtin_amdgcn_wmma_f32_16x16x32_f16(
              false, ahi, false, bhi, (short)0, acc[nt], false, false);
          acc[nt] = __builtin_amdgcn_wmma_f32_16x16x32_f16(
              false, ahi, false, blo, (short)0, acc[nt], false, false);
          acc[nt] = __builtin_amdgcn_wmma_f32_16x16x32_f16(
              false, alo, false, bhi, (short)0, acc[nt], false, false);
        }
      }

      // ---- store D (+bias): VGPR v -> row v (+8 for lanes>=16), lane&15 -> col
      const int rbase = tile * 16 + ((lane & 16) ? 8 : 0);
      #pragma unroll
      for (int nt = 0; nt < 4; ++nt) {
        int col = nt * 16 + (lane & 15);
        float bb = bs[m][col];
        #pragma unroll
        for (int v = 0; v < 8; ++v) {
          int r = rbase + v;
          if (r < n) outs[m][(size_t)r * DOUT + col] = acc[nt][v] + bb;
        }
      }
    }
  }
}

// ---------------- edge scatter: acc[dst] += xw[src] * dinv[src]*dinv[dst] --
// One wave per edge, each lane handles 2 columns (float2 gather, 4 atomics).
__global__ void __launch_bounds__(256)
k_scatter(const int* __restrict__ src, const int* __restrict__ dst,
          const float* __restrict__ dinv,
          const float* __restrict__ xw_hp, const float* __restrict__ xw_lp,
          float* __restrict__ acc_hp, float* __restrict__ acc_lp, int e)
{
  long long t = (long long)blockIdx.x * blockDim.x + threadIdx.x;
  int edge = (int)(t >> 5);
  int lane = (int)(t & 31);
  if (edge >= e) return;
  int s = src[edge];
  int d = dst[edge];
  float w = dinv[s] * dinv[d];
  int c0 = lane * 2;
  float2 vh = *(const float2*)(xw_hp + (size_t)s * DOUT + c0);
  float2 vl = *(const float2*)(xw_lp + (size_t)s * DOUT + c0);
  float* ph = acc_hp + (size_t)d * DOUT + c0;
  float* pl = acc_lp + (size_t)d * DOUT + c0;
  atomicAdd(ph,     vh.x * w);
  atomicAdd(ph + 1, vh.y * w);
  atomicAdd(pl,     vl.x * w);
  atomicAdd(pl + 1, vl.y * w);
}

// ---------------- per-node finalize: branches, gates, log_softmax ----------
__global__ void __launch_bounds__(256)
k_finalize(const float* __restrict__ xw_hp, const float* __restrict__ xw_lp,
           const float* __restrict__ xw_i,
           const float* __restrict__ acc_hp, const float* __restrict__ acc_lp,
           const float* __restrict__ dinv,
           const float* __restrict__ w_h, const float* __restrict__ b_h,
           const float* __restrict__ w_l, const float* __restrict__ b_l,
           const float* __restrict__ w_i2, const float* __restrict__ b_i2,
           float* __restrict__ out, int n)
{
  long long t = (long long)blockIdx.x * blockDim.x + threadIdx.x;
  int node = (int)(t >> 5);
  int lane = (int)(t & 31);
  if (node >= n) return;
  int c0 = lane * 2;
  size_t base = (size_t)node * DOUT + c0;

  float2 xh = *(const float2*)(xw_hp + base);
  float2 xl = *(const float2*)(xw_lp + base);
  float2 xi = *(const float2*)(xw_i  + base);
  float2 ph = *(const float2*)(acc_hp + base);
  float2 pl = *(const float2*)(acc_lp + base);
  float di = dinv[node];
  float sw = di * di;                       // self-loop norm

  float php0 = ph.x + xh.x * sw, php1 = ph.y + xh.y * sw;
  float plp0 = pl.x + xl.x * sw, plp1 = pl.y + xl.y * sw;

  float Hh0 = fmaxf(xh.x - php0, 0.0f), Hh1 = fmaxf(xh.y - php1, 0.0f);
  float Hl0 = fmaxf(plp0, 0.0f),        Hl1 = fmaxf(plp1, 0.0f);
  float Hi0 = fmaxf(xi.x, 0.0f),        Hi1 = fmaxf(xi.y, 0.0f);

  float dh = Hh0 * w_h[c0]  + Hh1 * w_h[c0 + 1];
  float dl = Hl0 * w_l[c0]  + Hl1 * w_l[c0 + 1];
  float dv = Hi0 * w_i2[c0] + Hi1 * w_i2[c0 + 1];
  dh = wave_sum(dh); dl = wave_sum(dl); dv = wave_sum(dv);

  float ah = 1.0f / (1.0f + expf(-(dh + b_h[0])));
  float al = 1.0f / (1.0f + expf(-(dl + b_l[0])));
  float ai = 1.0f / (1.0f + expf(-(dv + b_i2[0])));

  float o0 = ah * Hh0 + al * Hl0 + ai * Hi0;
  float o1 = ah * Hh1 + al * Hl1 + ai * Hi1;

  float mx  = wave_max(fmaxf(o0, o1));
  float se  = wave_sum(expf(o0 - mx) + expf(o1 - mx));
  float lse = mx + logf(se);

  float2 o; o.x = o0 - lse; o.y = o1 - lse;
  *(float2*)(out + base) = o;
}

extern "C" void kernel_launch(void* const* d_in, const int* in_sizes, int n_in,
                              void* d_out, int out_size, void* d_ws, size_t ws_size,
                              hipStream_t stream)
{
  const float* x    = (const float*)d_in[0];
  const int*   ei   = (const int*)d_in[1];
  const float* W_hp = (const float*)d_in[2];
  const float* b_hp = (const float*)d_in[3];
  const float* W_lp = (const float*)d_in[4];
  const float* b_lp = (const float*)d_in[5];
  const float* W_i  = (const float*)d_in[6];
  const float* b_i  = (const float*)d_in[7];
  const float* w_h  = (const float*)d_in[8];
  const float* b_h  = (const float*)d_in[9];
  const float* w_l  = (const float*)d_in[10];
  const float* b_l  = (const float*)d_in[11];
  const float* w_i2 = (const float*)d_in[12];
  const float* b_i2 = (const float*)d_in[13];
  float* out = (float*)d_out;

  const int n = in_sizes[0] / DIN;
  const int e = in_sizes[1] / 2;
  const int* src = ei;         // edge_index[0]
  const int* dst = ei + e;     // edge_index[1]

  float* f      = (float*)d_ws;
  float* deg    = f;
  float* dinv   = deg + n;
  float* xw_hp  = dinv + n;
  float* xw_lp  = xw_hp + (size_t)n * DOUT;
  float* xw_i   = xw_lp + (size_t)n * DOUT;
  float* acc_hp = xw_i  + (size_t)n * DOUT;
  float* acc_lp = acc_hp + (size_t)n * DOUT;

  {
    long long tot = (long long)n * DOUT;
    k_init<<<(int)((tot + 255) / 256), 256, 0, stream>>>(deg, acc_hp, acc_lp, n);
  }
  k_deg<<<(e + 255) / 256, 256, 0, stream>>>(dst, deg, e);
  k_dinv<<<(n + 255) / 256, 256, 0, stream>>>(deg, dinv, n);
  {
    int numTiles = (n + 15) / 16;
    int blocks   = (numTiles + 7) / 8;
    k_gemm<<<blocks, 256, 0, stream>>>(x, W_hp, b_hp, W_lp, b_lp, W_i, b_i,
                                       xw_hp, xw_lp, xw_i, n, numTiles);
  }
  {
    long long tot = (long long)e * 32;
    k_scatter<<<(int)((tot + 255) / 256), 256, 0, stream>>>(
        src, dst, dinv, xw_hp, xw_lp, acc_hp, acc_lp, e);
  }
  {
    long long tot = (long long)n * 32;
    k_finalize<<<(int)((tot + 255) / 256), 256, 0, stream>>>(
        xw_hp, xw_lp, xw_i, acc_hp, acc_lp, dinv,
        w_h, b_h, w_l, b_l, w_i2, b_i2, out, n);
  }
}
